// DecoupledCrossAttention_66821101191246
// MI455X (gfx1250) — compile-verified, matrix operand order checked by reading, not run on hardware
//
#include <hip/hip_runtime.h>
#include <math.h>

// ---------------------------------------------------------------------------
// DecoupledCrossAttention on MI455X (gfx1250, wave32, WMMA bf16 16x16x32)
//
// Pipeline:
//   f32->bf16 converts -> Q/K/V projections (WMMA GEMM, V stored transposed
//   per head) -> flash attention (block-cooperative: async-copy K/V tiles to
//   LDS, double buffered, 8 waves x 16 queries share them; transposed-score
//   trick keeps softmax lane-local) -> bf16 convert -> output projection.
// ---------------------------------------------------------------------------

typedef __bf16 bf16;
typedef __attribute__((ext_vector_type(16))) bf16 v16bf;
typedef __attribute__((ext_vector_type(8)))  bf16 v8bf;
typedef __attribute__((ext_vector_type(8)))  float v8f;

#define DIM   512
#define HEADS 8
#define HD    64
#define BATCH 4
#define NQ    2048
#define QB    128                    // queries per block (8 waves x 16)
#define SCALE 0.04419417382415922f   // 512^-0.5

__device__ __forceinline__ v8f wmma_bf16(v16bf a, v16bf b, v8f c) {
  // D = A(16x32) * B(32x16) + C, f32 accumulate
  return __builtin_amdgcn_wmma_f32_16x16x32_bf16(
      /*neg_a=*/false, a, /*neg_b=*/false, b,
      /*c_mod=*/(short)0, c, /*reuse_a=*/false, /*reuse_b=*/false);
}

// CDNA5 async global->LDS copy (ASYNCcnt path), 16 bytes per lane.
__device__ __forceinline__ void async_b128(unsigned lds_off, const bf16* gaddr) {
  asm volatile("global_load_async_to_lds_b128 %0, %1, off"
               :: "v"(lds_off), "v"(gaddr)
               : "memory");
}
__device__ __forceinline__ void wait_async() {
  asm volatile("s_wait_asynccnt 0" ::: "memory");
}

// Load a 16-bit A- or B-fragment from a K-major tile (global or LDS).
// A-frag: lane&15 = M row, chunks at kbase=(lane>>4)*8 : K=kb..kb+7, kb+16..kb+23
// B-frag (columns = rows of the given matrix): identical pattern by symmetry.
__device__ __forceinline__ v16bf load_frag(const bf16* __restrict__ base,
                                           int ld, int k0, int lane) {
  int row = lane & 15;
  int kb  = (lane >> 4) << 3;
  const bf16* p = base + (size_t)row * ld + k0 + kb;
  v8bf lo = *(const v8bf*)p;          // K = kb .. kb+7
  v8bf hi = *(const v8bf*)(p + 16);   // K = kb+16 .. kb+23
  v16bf r;
#pragma unroll
  for (int i = 0; i < 8; ++i) { r[i] = lo[i]; r[i + 8] = hi[i]; }
  return r;
}

__global__ void cvt_f32_to_bf16(const float* __restrict__ in,
                                bf16* __restrict__ out, int n) {
  int i = blockIdx.x * blockDim.x + threadIdx.x;
  int stride = gridDim.x * blockDim.x;
  for (; i < n; i += stride) out[i] = (bf16)in[i];
}

// C = A(MxK) * W(NxK)^T + bias
// mode 0: bf16 row-major C[M x N]
// mode 1: bf16 transposed-per-head Vt[b, h, d, m_local]  (ctx_len = rows/batch)
// mode 2: f32 row-major C[M x N]
__global__ void gemm_wmma(const bf16* __restrict__ A, const bf16* __restrict__ W,
                          const float* __restrict__ bias, void* __restrict__ C,
                          int M, int N, int K, int mode, int ctx_len) {
  int lane = threadIdx.x & 31;
  int wid  = (blockIdx.x * blockDim.x + threadIdx.x) >> 5;
  int ntg  = N >> 6;                 // 64-column tile groups
  int tM   = wid / ntg;
  int tN   = wid % ntg;
  if (tM * 16 >= M) return;

  const bf16* Abase = A + (size_t)tM * 16 * K;
  int n0 = tN * 64;
  v8f acc[4] = {};

  for (int k0 = 0; k0 < K; k0 += 32) {
    v16bf af = load_frag(Abase, K, k0, lane);
#pragma unroll
    for (int t = 0; t < 4; ++t) {
      v16bf wf = load_frag(W + (size_t)(n0 + t * 16) * K, K, k0, lane);
      acc[t] = wmma_bf16(af, wf, acc[t]);
    }
  }

  int nl  = lane & 15;
  int mhi = (lane >> 4) << 3;
#pragma unroll
  for (int t = 0; t < 4; ++t) {
#pragma unroll
    for (int v = 0; v < 8; ++v) {
      int n = n0 + t * 16 + nl;
      int m = tM * 16 + v + mhi;
      float val = acc[t][v] + bias[n];
      if (mode == 0) {
        ((bf16*)C)[(size_t)m * N + n] = (bf16)val;
      } else if (mode == 2) {
        ((float*)C)[(size_t)m * N + n] = val;
      } else {  // transposed-per-head V
        int bb = m / ctx_len, ml = m % ctx_len;
        int h = n >> 6, d = n & 63;
        ((bf16*)C)[(((size_t)bb * HEADS + h) * HD + d) * ctx_len + ml] = (bf16)val;
      }
    }
  }
}

// Flash attention, block-cooperative: one block = one (b,h), 8 waves x 16
// queries.  K tile (32x64) and V^T tile (64x32) are async-copied to LDS,
// double buffered, and shared by all waves.
// S^T = K * Q^T  => lane = query, vgpr = key  => lane-local softmax + trivial
// repack of P into the B-fragment for O^T = V^T * P.
__global__ void __launch_bounds__(256)
attn_wmma(const bf16* __restrict__ Q, const bf16* __restrict__ Kc,
          const bf16* __restrict__ Vt, float* __restrict__ Out,
          int Mctx, int accumulate) {
  __shared__ __align__(16) bf16 Ksh[2][32][64];   // [buf][key][d]
  __shared__ __align__(16) bf16 Vsh[2][64][32];   // [buf][d][m]

  int t    = threadIdx.x;
  int lane = t & 31;
  int wv   = t >> 5;                       // wave in block 0..7
  int qblk = blockIdx.x % (NQ / QB);
  int bh   = blockIdx.x / (NQ / QB);
  int b = bh / HEADS, h = bh % HEADS;

  const bf16* Qbase = Q  + ((size_t)(b * NQ + qblk * QB + wv * 16) * DIM) + h * HD;
  const bf16* Kbase = Kc + ((size_t)b * Mctx * DIM) + h * HD;
  const bf16* Vbase = Vt + ((size_t)(b * HEADS + h) * HD) * Mctx;

  // Hoisted loop-invariant Q B-fragments (d0 = 0 and 32).
  v16bf qf0 = load_frag(Qbase, DIM, 0, lane);
  v16bf qf1 = load_frag(Qbase, DIM, 32, lane);

  // Per-thread staging slots: 256 threads x 16B cover each 4KB tile exactly.
  int krow = t >> 3, kcol = (t & 7) * 8;   // K: 32 rows x 64 d
  int vrow = t >> 2, vcol = (t & 3) * 8;   // V: 64 rows x 32 m

  float m_run = -1e30f, l_run = 0.f;
  v8f o[4] = {};

  const int nb = Mctx / 32;

  // Prologue: stage buffer 0 for key block 0.
  async_b128((unsigned)(size_t)&Ksh[0][krow][kcol],
             Kbase + (size_t)krow * DIM + kcol);
  async_b128((unsigned)(size_t)&Vsh[0][vrow][vcol],
             Vbase + (size_t)vrow * Mctx + vcol);

  for (int ib = 0; ib < nb; ++ib) {
    int cur = ib & 1;
    wait_async();        // this thread's copies into buf[cur] have landed
    __syncthreads();     // everyone's copies landed; prev compute also done

    if (ib + 1 < nb) {   // prefetch next key block into the other buffer
      int kb2 = (ib + 1) * 32;
      async_b128((unsigned)(size_t)&Ksh[cur ^ 1][krow][kcol],
                 Kbase + (size_t)(kb2 + krow) * DIM + kcol);
      async_b128((unsigned)(size_t)&Vsh[cur ^ 1][vrow][vcol],
                 Vbase + (size_t)vrow * Mctx + kb2 + vcol);
    }

    // ---- S^T tiles: keys 0..15 (s0) and 16..31 (s1) of this block ----
    v8f s0 = {}, s1 = {};
    {
      v16bf k00 = load_frag(&Ksh[cur][0][0],  64, 0,  lane);
      v16bf k01 = load_frag(&Ksh[cur][0][0],  64, 32, lane);
      v16bf k10 = load_frag(&Ksh[cur][16][0], 64, 0,  lane);
      v16bf k11 = load_frag(&Ksh[cur][16][0], 64, 32, lane);
      s0 = wmma_bf16(k00, qf0, s0);
      s0 = wmma_bf16(k01, qf1, s0);
      s1 = wmma_bf16(k10, qf0, s1);
      s1 = wmma_bf16(k11, qf1, s1);
    }

    // ---- online softmax (lane = query; keys split lane vs lane^16) ----
    float mx = -1e30f;
#pragma unroll
    for (int v = 0; v < 8; ++v) {
      s0[v] *= SCALE; s1[v] *= SCALE;
      mx = fmaxf(mx, fmaxf(s0[v], s1[v]));
    }
    mx = fmaxf(mx, __shfl_xor(mx, 16, 32));
    float m_new = fmaxf(m_run, mx);
    float corr  = __expf(m_run - m_new);

    float psum = 0.f;
#pragma unroll
    for (int v = 0; v < 8; ++v) {
      float p0 = __expf(s0[v] - m_new);
      float p1 = __expf(s1[v] - m_new);
      s0[v] = p0; s1[v] = p1;
      psum += p0 + p1;
    }
    psum += __shfl_xor(psum, 16, 32);
    l_run = l_run * corr + psum;
    m_run = m_new;

    // P -> B-fragment (lane-local; C-layout vgpr/key order matches K-packing)
    v16bf pB;
#pragma unroll
    for (int e = 0; e < 8; ++e) { pB[e] = (bf16)s0[e]; pB[e + 8] = (bf16)s1[e]; }

    // ---- O^T += V^T * P ----
#pragma unroll
    for (int dt = 0; dt < 4; ++dt) {
      v16bf vf = load_frag(&Vsh[cur][dt * 16][0], 32, 0, lane);
#pragma unroll
      for (int v = 0; v < 8; ++v) o[dt][v] *= corr;
      o[dt] = wmma_bf16(vf, pB, o[dt]);   // vgpr = d, lane = query
    }
  }

  float rl = 1.f / l_run;
  int i   = lane & 15;
  int mhi = (lane >> 4) << 3;
#pragma unroll
  for (int dt = 0; dt < 4; ++dt) {
#pragma unroll
    for (int v = 0; v < 8; ++v) {
      int d = dt * 16 + v + mhi;
      float val = o[dt][v] * rl;
      float* dst = Out + ((size_t)(b * NQ + qblk * QB + wv * 16 + i) * DIM) + h * HD + d;
      if (accumulate) *dst += val; else *dst = val;
    }
  }
}

extern "C" void kernel_launch(void* const* d_in, const int* in_sizes, int n_in,
                              void* d_out, int out_size, void* d_ws, size_t ws_size,
                              hipStream_t stream) {
  const float* x      = (const float*)d_in[0];
  const float* audio  = (const float*)d_in[1];
  const float* singer = (const float*)d_in[2];
  const float* Wq  = (const float*)d_in[3];  const float* bq  = (const float*)d_in[4];
  const float* Wka = (const float*)d_in[5];  const float* bka = (const float*)d_in[6];
  const float* Wva = (const float*)d_in[7];  const float* bva = (const float*)d_in[8];
  const float* Wks = (const float*)d_in[9];  const float* bks = (const float*)d_in[10];
  const float* Wvs = (const float*)d_in[11]; const float* bvs = (const float*)d_in[12];
  const float* Wp  = (const float*)d_in[13]; const float* bp  = (const float*)d_in[14];

  const int n_x = BATCH * NQ * DIM;        // 4,194,304
  const int n_a = BATCH * 2048 * DIM;      // audio ctx
  const int n_s = BATCH * 256 * DIM;       // singer ctx
  const int n_w = DIM * DIM;

  char* ws = (char*)d_ws;
  size_t off = 0;
  auto alloc = [&](size_t bytes) -> void* {
    off = (off + 255) & ~(size_t)255;
    void* p = ws + off;
    off += bytes;
    return p;
  };

  bf16* xb   = (bf16*)alloc((size_t)n_x * 2);
  bf16* ab   = (bf16*)alloc((size_t)n_a * 2);
  bf16* sb   = (bf16*)alloc((size_t)n_s * 2);
  bf16* wqb  = (bf16*)alloc((size_t)n_w * 2);
  bf16* wkab = (bf16*)alloc((size_t)n_w * 2);
  bf16* wvab = (bf16*)alloc((size_t)n_w * 2);
  bf16* wksb = (bf16*)alloc((size_t)n_w * 2);
  bf16* wvsb = (bf16*)alloc((size_t)n_w * 2);
  bf16* wpb  = (bf16*)alloc((size_t)n_w * 2);
  bf16* qb   = (bf16*)alloc((size_t)n_x * 2);
  bf16* kab  = (bf16*)alloc((size_t)n_a * 2);
  bf16* vat  = (bf16*)alloc((size_t)n_a * 2);   // V audio, transposed per head
  bf16* ksb  = (bf16*)alloc((size_t)n_s * 2);
  bf16* vst  = (bf16*)alloc((size_t)n_s * 2);   // V singer, transposed per head
  float* attnf = (float*)alloc((size_t)n_x * 4);
  bf16* attnb  = (bf16*)alloc((size_t)n_x * 2);

  auto cvt = [&](const float* in, bf16* out, int n) {
    int blocks = (n + 255) / 256;
    cvt_f32_to_bf16<<<blocks, 256, 0, stream>>>(in, out, n);
  };
  auto gemm = [&](const bf16* A, const bf16* W, const float* bias, void* C,
                  int M, int N, int K, int mode, int ctx) {
    int waves  = (M / 16) * (N / 64);
    int blocks = (waves + 7) / 8;   // 8 waves / 256-thread block
    gemm_wmma<<<blocks, 256, 0, stream>>>(A, W, bias, C, M, N, K, mode, ctx);
  };

  // --- precision convert ---
  cvt(x, xb, n_x);  cvt(audio, ab, n_a);  cvt(singer, sb, n_s);
  cvt(Wq, wqb, n_w);  cvt(Wka, wkab, n_w); cvt(Wva, wvab, n_w);
  cvt(Wks, wksb, n_w); cvt(Wvs, wvsb, n_w); cvt(Wp, wpb, n_w);

  // --- projections ---
  gemm(xb, wqb,  bq,  qb,  BATCH * NQ,   DIM, DIM, 0, 0);
  gemm(ab, wkab, bka, kab, BATCH * 2048, DIM, DIM, 0, 0);
  gemm(ab, wvab, bva, vat, BATCH * 2048, DIM, DIM, 1, 2048);
  gemm(sb, wksb, bks, ksb, BATCH * 256,  DIM, DIM, 0, 0);
  gemm(sb, wvsb, bvs, vst, BATCH * 256,  DIM, DIM, 1, 256);

  // --- attention: audio (store) then singer (accumulate, lambda=1) ---
  {
    int blocks = BATCH * HEADS * (NQ / QB);   // 512 blocks, 256 threads
    attn_wmma<<<blocks, 256, 0, stream>>>(qb, kab, vat, attnf, 2048, 0);
    attn_wmma<<<blocks, 256, 0, stream>>>(qb, ksb, vst, attnf, 256, 1);
  }

  // --- output projection -> f32 d_out ---
  cvt(attnf, attnb, n_x);
  gemm(attnb, wpb, bp, d_out, BATCH * NQ, DIM, DIM, 2, 0);
}
